// SymSATNet_basis_36189394436511
// MI455X (gfx1250) — matrix operand order repaired and checked by reading, not compiled, not run on hardware
//
#include <hip/hip_runtime.h>
#include <hip/hip_bf16.h>

// ---------------------------------------------------------------------------
// SymSATNet mixer for MI455X (gfx1250, wave32).
//
//  K1 build_C : C (730x730 fp32, padded 736x736) = weighted basis sum +
//               border; also Ct16 = f16 transpose(C) (WMMA A-operand).
//               Bandwidth bound: 85 MB basis read ~= 4 us @ 23.3 TB/s.
//  K2 mix     : one workgroup per batch (8 blocks x 512 threads = 16 waves).
//               V, G = C^T V (f32) and a f16 copy of V live in LDS (~248 KB
//               of the 320 KB WGP LDS). G is initialized with
//               v_wmma_f32_16x16x32_f16 tiles; the 7290 sequential coordinate
//               steps then use the incremental form:
//                 g = G[i] - C[i,i]*V[i];  vnew = -g/||g||;
//                 G += C[i,:] outer (vnew - V[i])        (parallel rank-1)
//               The NEXT permuted C row is double-buffered into LDS with
//               GLOBAL_LOAD_ASYNC_TO_LDS_B128 (ASYNCcnt) issued by wave 1
//               during the current step's update, so the serial loop never
//               waits on global memory.
// Workspace: Cg 736*736*4 + Ct16 736*736*2 ~= 3.25 MB.
// ---------------------------------------------------------------------------

#define NN      729
#define NP1     730
#define PAD     736          // 46*16, multiple of WMMA tile; zero padded
#define KD      32
#define NBASIS  40
#define MAXIT   10
#define NTHREADS 512
#define NWAVES  (NTHREADS / 32)
#define PI_F    3.14159265358979323846f

typedef __attribute__((ext_vector_type(16))) _Float16 v16h;
typedef __attribute__((ext_vector_type(8)))  _Float16 v8h;
typedef __attribute__((ext_vector_type(8)))  float    v8f;

__device__ __forceinline__ float wred32(float x) {
  #pragma unroll
  for (int o = 16; o > 0; o >>= 1) x += __shfl_xor(x, o, 32);
  return x;
}

// Flat address of a __shared__ object = {LDS aperture, 32-bit LDS offset}
// (ISA 10.2 aperture mapping), so the low 32 bits are the LDS byte address
// needed by the async-copy VDST operand.
__device__ __forceinline__ unsigned lds_off(const void* p) {
  return (unsigned)(unsigned long long)p;
}

// One wave stages 6 x (32 lanes x 16B) = 3072 B >= one 2944 B C row.
// offset: applies to BOTH the LDS dest and the global source (ISA 15.18.3).
#define ASYNC_ROW_CHUNK(off)                                                  \
  asm volatile("global_load_async_to_lds_b128 %0, %1, off offset:" #off       \
               :: "v"(dstv), "v"(srcv) : "memory")

// --------------------------------------------------------------------------
// K1: build padded C (fp32, row-major, ld=PAD) and Ct16[j][i] = (f16)C[i][j].
// --------------------------------------------------------------------------
__global__ void build_C_kernel(const float* __restrict__ coeff,
                               const float* __restrict__ upper,
                               const float* __restrict__ basis,
                               float* __restrict__ Cg,
                               _Float16* __restrict__ Ct) {
  __shared__ float cf[NBASIS];
  const int t = threadIdx.x;
  if (t < NBASIS) cf[t] = coeff[t];
  __syncthreads();

  const int idx = blockIdx.x * blockDim.x + t;
  if (idx >= PAD * PAD) return;
  const int i = idx / PAD;
  const int j = idx - i * PAD;

  float val = 0.0f;
  if (i < NP1 && j < NP1) {
    if (i == 0 && j == 0) {
      val = 0.0f;
    } else if (i == 0) {
      val = upper[j - 1];
    } else if (j == 0) {
      val = upper[i - 1];
    } else {
      const float* bp = basis + (size_t)(i - 1) * NN + (j - 1);
      #pragma unroll 4
      for (int b = 0; b < NBASIS; ++b)
        val = fmaf(cf[b], bp[(size_t)b * NN * NN], val);
    }
  }
  Cg[(size_t)i * PAD + j] = val;
  Ct[(size_t)j * PAD + i] = (_Float16)val;   // transposed copy for WMMA A
}

// --------------------------------------------------------------------------
// K2: per-batch init-V, WMMA init of G = C^T V, sequential mixing, output.
// --------------------------------------------------------------------------
__global__ __launch_bounds__(NTHREADS)
void mix_kernel(const float* __restrict__ Cg,
                const _Float16* __restrict__ Ct,
                const float* __restrict__ z,
                const int* __restrict__ is_input,
                const float* __restrict__ vraw,
                const int* __restrict__ perm,
                float* __restrict__ out) {
  extern __shared__ float smem[];
  float*    Vs    = smem;                          // PAD*KD f32
  float*    Gs    = Vs + PAD * KD;                 // PAD*KD f32
  _Float16* Vh    = (_Float16*)(Gs + PAD * KD);    // PAD*KD f16 (B panel)
  float*    diagS = (float*)(Vh + PAD * KD);       // PAD
  float*    zS    = diagS + PAD;                   // PAD (z_full)
  int*      permS = (int*)(zS + PAD);              // PAD
  int*      freeS = permS + PAD;                   // PAD
  float*    row0  = (float*)(freeS + PAD);         // 768 (async C-row buf 0)
  float*    row1  = row0 + 768;                    // 768 (async C-row buf 1)
  float*    dlt   = row1 + 768;                    // KD
  int*      skipF = (int*)(dlt + KD);              // 1

  const int b    = blockIdx.x;
  const int tid  = threadIdx.x;
  const int lane = tid & 31;
  const int w    = tid >> 5;

  // ---- stage small per-batch tables -------------------------------------
  for (int i = tid; i < PAD; i += NTHREADS) {
    diagS[i] = Cg[(size_t)i * PAD + i];
    const bool inner = (i >= 1) && (i < NP1);
    zS[i]    = inner ? z[(size_t)b * NN + i - 1] : 1.0f;
    freeS[i] = inner ? (is_input[(size_t)b * NN + i - 1] == 0) : 0;
    permS[i] = (i < NN) ? perm[i] : 1;
  }
  // zero padded V rows (730..735) so WMMA K-tail contributes 0
  for (int idx = tid; idx < (PAD - NP1) * KD; idx += NTHREADS)
    Vs[NP1 * KD + idx] = 0.0f;

  // ---- init V row 0: v0 = normalize(v_raw[b,0]) -------------------------
  if (tid < KD) {
    float r  = vraw[(size_t)b * NP1 * KD + tid];
    float nn = sqrtf(wred32(r * r));
    Vs[tid] = r / nn;
  }
  __syncthreads();

  // ---- init V rows 1..729 (one wave per row) ----------------------------
  for (int n = 1 + w; n < NP1; n += NWAVES) {
    float r   = vraw[((size_t)b * NP1 + n) * KD + lane];
    float v0k = Vs[lane];
    float dot = wred32(r * v0k);
    float u   = r - dot * v0k;
    float nu  = fmaxf(sqrtf(wred32(u * u)), 1e-8f);
    u /= nu;
    float zf = zS[n];
    Vs[n * KD + lane] = -__cosf(PI_F * zf) * v0k + __sinf(PI_F * zf) * u;
  }
  __syncthreads();

  // ---- f16 panel of V for the WMMA B operand ----------------------------
  for (int idx = tid; idx < PAD * KD; idx += NTHREADS)
    Vh[idx] = (_Float16)Vs[idx];
  __syncthreads();

  // ---- G = C^T V via v_wmma_f32_16x16x32_f16 ----------------------------
  // A (16x32 f16): lane m=lane&15; lanes<16 hold K 0..7 & 16..23,
  //                lanes>=16 hold K 8..15 & 24..31 (ISA 16-bit A layout).
  // B (32x16 f16): lane L holds column N=L&15, K in the same order as A.
  // D (16x16 f32): VGPR r -> M = r + 8*(lane>=16), N = lane&15.
  {
    const int m     = lane & 15;
    const int jbase = (lane < 16) ? 0 : 8;
    for (int tile = w; tile < (PAD / 16) * 2; tile += NWAVES) {
      const int i0 = (tile >> 1) * 16;
      const int k0 = (tile & 1) * 16;
      v8f acc = {};
      const _Float16* arow = Ct + (size_t)(i0 + m) * PAD;
      for (int j0 = 0; j0 < PAD; j0 += 32) {
        v8h lo = *(const v8h*)(arow + j0 + jbase);
        v8h hi = *(const v8h*)(arow + j0 + jbase + 16);
        v16h a;
        #pragma unroll
        for (int t2 = 0; t2 < 8; ++t2) { a[t2] = lo[t2]; a[t2 + 8] = hi[t2]; }
        v16h bb;
        #pragma unroll
        for (int t2 = 0; t2 < 8; ++t2) {
          bb[t2]     = Vh[(j0 + jbase + t2) * KD + k0 + m];
          bb[t2 + 8] = Vh[(j0 + jbase + 16 + t2) * KD + k0 + m];
        }
        acc = __builtin_amdgcn_wmma_f32_16x16x32_f16(
            false, a, false, bb, (short)0, acc, false, false);
      }
      const int rbase = i0 + ((lane >> 4) << 3);
      const int kk    = k0 + (lane & 15);
      #pragma unroll
      for (int r = 0; r < 8; ++r) Gs[(rbase + r) * KD + kk] = acc[r];
    }
  }
  __syncthreads();

  const unsigned rb[2] = { lds_off(row0), lds_off(row1) };

  // prologue: async-stage the first permuted C row into buffer 0
  if (w == 1) {
    const unsigned dstv = rb[0] + (unsigned)lane * 16u;
    unsigned long long srcv =
        (unsigned long long)(Cg + (size_t)permS[0] * PAD) + (unsigned)lane * 16u;
    ASYNC_ROW_CHUNK(0);    ASYNC_ROW_CHUNK(512);  ASYNC_ROW_CHUNK(1024);
    ASYNC_ROW_CHUNK(1536); ASYNC_ROW_CHUNK(2048); ASYNC_ROW_CHUNK(2560);
  }

  // ---- sequential mixing: 10 sweeps x 729 coordinate steps --------------
  int par = 0;
  for (int it = 0; it < MAXIT; ++it) {
    for (int s = 0; s < NN; ++s) {
      const int i = permS[s];

      if (tid < KD) {
        const float vi = Vs[i * KD + tid];
        const float g  = Gs[i * KD + tid] - diagS[i] * vi;
        const float n  = fmaxf(sqrtf(wred32(g * g)), 1e-8f);
        const float vnew = -g / n;
        const int   fr = freeS[i];
        const float d  = fr ? (vnew - vi) : 0.0f;
        if (fr) Vs[i * KD + tid] = vnew;
        dlt[tid] = d;
        if (tid == 0) *skipF = (fr == 0);
      } else if (w == 1) {
        // row for this step (issued last step) must have landed in LDS
        asm volatile("s_wait_asynccnt 0x0" ::: "memory");
      }
      __syncthreads();

      // overlap: wave 1 stages the NEXT row while everyone does the update
      if (w == 1) {
        const int nexti = permS[(s + 1 < NN) ? s + 1 : 0];  // perm repeats
        const unsigned dstv = rb[par ^ 1] + (unsigned)lane * 16u;
        unsigned long long srcv =
            (unsigned long long)(Cg + (size_t)nexti * PAD) + (unsigned)lane * 16u;
        ASYNC_ROW_CHUNK(0);    ASYNC_ROW_CHUNK(512);  ASYNC_ROW_CHUNK(1024);
        ASYNC_ROW_CHUNK(1536); ASYNC_ROW_CHUNK(2048); ASYNC_ROW_CHUNK(2560);
      }

      if (!*skipF) {
        // rank-1 update: G[j,k] += C[i,j] * delta[k]; row from LDS buffer
        const float* crow = par ? row1 : row0;
        const float dk = dlt[lane];
        #pragma unroll 2
        for (int j = w; j < PAD; j += NWAVES)
          Gs[j * KD + lane] = fmaf(crow[j], dk, Gs[j * KD + lane]);
      }
      __syncthreads();
      par ^= 1;
    }
  }

  // ---- output: z_out = ii ? z : acos(clip(-V.v0))/pi --------------------
  for (int i = 1 + tid; i < NP1; i += NTHREADS) {
    float dot = 0.0f;
    #pragma unroll
    for (int k = 0; k < KD; ++k)
      dot = fmaf(Vs[i * KD + k], Vs[k], dot);
    float ca = fminf(fmaxf(-dot, -1.0f + 1e-6f), 1.0f - 1e-6f);
    float zo = acosf(ca) * (1.0f / PI_F);
    out[(size_t)b * NN + i - 1] = freeS[i] ? zo : zS[i];
  }
}

// --------------------------------------------------------------------------
extern "C" void kernel_launch(void* const* d_in, const int* in_sizes, int n_in,
                              void* d_out, int out_size, void* d_ws, size_t ws_size,
                              hipStream_t stream) {
  const float* coeff    = (const float*)d_in[0];
  const float* upper    = (const float*)d_in[1];
  const float* basis    = (const float*)d_in[2];
  const float* z        = (const float*)d_in[3];
  const int*   is_input = (const int*)d_in[4];
  const float* vraw     = (const float*)d_in[5];
  const int*   perm     = (const int*)d_in[6];
  float* out = (float*)d_out;

  float*    Cg = (float*)d_ws;
  _Float16* Ct = (_Float16*)((char*)d_ws + (size_t)PAD * PAD * sizeof(float));

  const int nelems = PAD * PAD;
  build_C_kernel<<<(nelems + 255) / 256, 256, 0, stream>>>(coeff, upper, basis, Cg, Ct);

  // Vs + Gs (f32) + Vh (f16) + 4 tables + 2 row buffers + delta + flag
  const size_t ldsBytes =
      (size_t)(PAD * KD * 2 + PAD * KD / 2 + PAD * 4 + 2 * 768 + KD + 8) *
      sizeof(float);  // ~248 KB
  mix_kernel<<<8, NTHREADS, ldsBytes, stream>>>(Cg, Ct, z, is_input, vraw, perm, out);
}
